// FM_23991687315885
// MI455X (gfx1250) — matrix operand order repaired and checked by reading, not compiled, not run on hardware
//
#include <hip/hip_runtime.h>
#include <math.h>

typedef _Float16 half_t;
typedef __attribute__((ext_vector_type(16))) _Float16 v16h;
typedef __attribute__((ext_vector_type(8)))  _Float16 v8h;
typedef __attribute__((ext_vector_type(8)))  float    v8f;

#define DEVFN static __device__ __forceinline__

DEVFN float f_silu(float x){ return x / (1.f + __expf(-x)); }
DEVFN float f_softplus(float x){ return (x > 20.f) ? x : log1pf(__expf(x)); }
DEVFN float f_gelu(float x){ return 0.5f * x * (1.f + erff(x * 0.7071067811865475f)); }

// =============== WMMA GEMM: C[M,N] = A[M,K](f16) * W[N,K]^T(f16) + bias ===============
// One wave -> one 16x16 output tile, k-loop in steps of 32 via v_wmma_f32_16x16x32_f16.
// A fragment (16x32 f16): lane m=lane&15; elems 0..15 = K(kb..kb+15),
//   kb = 16*(lane>>4).  B fragment (32x16): lane n=lane&15; elems 0..15 = K(kh..kh+15),
//   kh = 16*(lane>>4).  C/D: elem e -> row tm + e + 8*(lane>>4), col tn + (lane&15).
__global__ void k_gemm(const half_t* __restrict__ A, const half_t* __restrict__ Wt,
                       const float* __restrict__ bias, float* __restrict__ Cout,
                       int N, int K, int ntiles)
{
    int wave = blockIdx.x * (blockDim.x >> 5) + (threadIdx.x >> 5);
    if (wave >= ntiles) return;                      // wave-uniform exit (EXEC stays full)
    int tncnt = N >> 4;
    int tm = (wave / tncnt) << 4;
    int tn = (wave % tncnt) << 4;
    int lane = threadIdx.x & 31;
    int hi = lane >> 4;
    int ln = lane & 15;
    const half_t* ap = A  + (size_t)(tm + ln) * K + (hi << 4);
    const half_t* bp = Wt + (size_t)(tn + ln) * K + (hi << 4);
    v8f acc = {};
    for (int k0 = 0; k0 < K; k0 += 32) {
        v8h alo = *(const v8h*)(ap + k0);
        v8h ahi = *(const v8h*)(ap + k0 + 8);
        v16h a, b;
        #pragma unroll
        for (int i = 0; i < 8; ++i) { a[i] = alo[i]; a[i + 8] = ahi[i]; }
        b = *(const v16h*)(bp + k0);
        acc = __builtin_amdgcn_wmma_f32_16x16x32_f16(false, a, false, b,
                                                     (short)0, acc, false, false);
    }
    int n = tn + ln;
    float bv = bias ? bias[n] : 0.f;
    int mbase = tm + (hi << 3);
    #pragma unroll
    for (int e = 0; e < 8; ++e)
        Cout[(size_t)(mbase + e) * N + n] = acc[e] + bv;
}

// =============== elementwise helpers ===============
__global__ void k_cvt(const float* __restrict__ s, half_t* __restrict__ d, int n)
{
    int t = blockIdx.x * blockDim.x + threadIdx.x;
    if (t < n) d[t] = (half_t)s[t];
}

// x_proj_w [4][36][128] -> f16 [4][48][128] zero padded
__global__ void k_pad_xpw(const float* __restrict__ s, half_t* __restrict__ d)
{
    int t = blockIdx.x * blockDim.x + threadIdx.x;
    if (t >= 4 * 48 * 128) return;
    int dd = t & 127; int c = (t >> 7) % 48; int k = t / (48 * 128);
    d[t] = (c < 36) ? (half_t)s[(size_t)k * 36 * 128 + c * 128 + dd] : (half_t)0.f;
}

// =============== stage 1: LN stats over C=128 (NCHW) ===============
__global__ void k_ln_stats(const float* __restrict__ x, float* __restrict__ mean,
                           float* __restrict__ rstd)
{
    int p = blockIdx.x * blockDim.x + threadIdx.x;   // b*16384 + hw
    if (p >= 65536) return;
    int b = p >> 14, hw = p & 16383;
    const float* xp = x + (size_t)b * 128 * 16384 + hw;
    float s = 0.f, s2 = 0.f;
    for (int c = 0; c < 128; ++c) { float v = xp[(size_t)c * 16384]; s += v; s2 += v * v; }
    float m = s * (1.f / 128.f);
    mean[p] = m;
    rstd[p] = rsqrtf(s2 * (1.f / 128.f) - m * m + 1e-5f);
}

// maxpool(x)->inp ; LN values -> xn1 (c<64 full res, NCHW) and x2p (c>=64, maxpooled)
__global__ void k_prep1(const float* __restrict__ x, const float* __restrict__ mean,
                        const float* __restrict__ rstd, const float* __restrict__ g,
                        const float* __restrict__ bta, float* __restrict__ inp,
                        float* __restrict__ xn1, float* __restrict__ x2p)
{
    int t = blockIdx.x * blockDim.x + threadIdx.x;   // b*128*4096 + c*4096 + p2
    if (t >= 4 * 128 * 4096) return;
    int p2 = t & 4095; int c = (t >> 12) & 127; int b = t >> 19;
    int h2 = p2 >> 6, w2 = p2 & 63;
    size_t base = (size_t)(b * 128 + c) * 16384;
    int hw = (2 * h2) * 128 + 2 * w2;
    float v00 = x[base + hw],        v01 = x[base + hw + 1];
    float v10 = x[base + hw + 128],  v11 = x[base + hw + 129];
    inp[t] = fmaxf(fmaxf(v00, v01), fmaxf(v10, v11));
    size_t pb = (size_t)b * 16384;
    float gc = g[c], bc = bta[c];
    float n00 = (v00 - mean[pb + hw])       * rstd[pb + hw]       * gc + bc;
    float n01 = (v01 - mean[pb + hw + 1])   * rstd[pb + hw + 1]   * gc + bc;
    float n10 = (v10 - mean[pb + hw + 128]) * rstd[pb + hw + 128] * gc + bc;
    float n11 = (v11 - mean[pb + hw + 129]) * rstd[pb + hw + 129] * gc + bc;
    if (c < 64) {
        size_t ob = (size_t)(b * 64 + c) * 16384;
        xn1[ob + hw] = n00; xn1[ob + hw + 1] = n01;
        xn1[ob + hw + 128] = n10; xn1[ob + hw + 129] = n11;
    } else {
        x2p[(size_t)(b * 64 + (c - 64)) * 4096 + p2] = fmaxf(fmaxf(n00, n01), fmaxf(n10, n11));
    }
}

// im2col pack for patch-embed 2x2/stride2: Ape[m=(b,h2,w2)][k=i*4+kh*2+kw]
__global__ void k_pack_peA(const float* __restrict__ xn1, half_t* __restrict__ Ape)
{
    int t = blockIdx.x * blockDim.x + threadIdx.x;   // m*64 + i
    if (t >= 16384 * 64) return;
    int i = t & 63; int m = t >> 6;
    int b = m >> 12; int p2 = m & 4095; int h2 = p2 >> 6, w2 = p2 & 63;
    size_t src = (size_t)(b * 64 + i) * 16384 + (2 * h2) * 128 + 2 * w2;
    size_t dst = (size_t)m * 256 + i * 4;
    Ape[dst + 0] = (half_t)xn1[src];
    Ape[dst + 1] = (half_t)xn1[src + 1];
    Ape[dst + 2] = (half_t)xn1[src + 128];
    Ape[dst + 3] = (half_t)xn1[src + 129];
}

// pe_ln -> x1 (residual base, [16384,64] f32), then vss_ln -> x1n f16
__global__ void k_pe_ln(const float* __restrict__ peout,
                        const float* __restrict__ g1, const float* __restrict__ b1,
                        const float* __restrict__ g2, const float* __restrict__ b2,
                        float* x1, half_t* __restrict__ x1n)
{
    int m = blockIdx.x * blockDim.x + threadIdx.x;
    if (m >= 16384) return;
    const float* r = peout + (size_t)m * 64;
    float s = 0.f, s2 = 0.f;
    for (int c = 0; c < 64; ++c) { float v = r[c]; s += v; s2 += v * v; }
    float mu = s * (1.f / 64.f), rs = rsqrtf(s2 * (1.f / 64.f) - mu * mu + 1e-5f);
    float t1 = 0.f, t2 = 0.f;
    for (int c = 0; c < 64; ++c) {
        float v = (r[c] - mu) * rs * g1[c] + b1[c];
        x1[(size_t)m * 64 + c] = v; t1 += v; t2 += v * v;
    }
    float mu2 = t1 * (1.f / 64.f), rs2 = rsqrtf(t2 * (1.f / 64.f) - mu2 * mu2 + 1e-5f);
    for (int c = 0; c < 64; ++c) {
        float v = x1[(size_t)m * 64 + c];
        x1n[(size_t)m * 64 + c] = (half_t)((v - mu2) * rs2 * g2[c] + b2[c]);
    }
}

// depthwise 3x3 + SiLU on xc (= xz[:, :128], pixel-major); emit f32 pixel-major + f16 hw/wh
__global__ void k_dw_silu(const float* __restrict__ xz, const float* __restrict__ w,
                          const float* __restrict__ bias, float* __restrict__ xcpm,
                          half_t* __restrict__ Ahw, half_t* __restrict__ Awh)
{
    int t = blockIdx.x * blockDim.x + threadIdx.x;   // b*4096*128 + p*128 + d
    if (t >= 4 * 4096 * 128) return;
    int d = t & 127; int p = (t >> 7) & 4095; int b = t >> 19;
    int h = p >> 6, wq = p & 63;
    float acc = bias[d];
    #pragma unroll
    for (int ky = 0; ky < 3; ++ky) {
        int hh = h + ky - 1; if (hh < 0 || hh > 63) continue;
        #pragma unroll
        for (int kx = 0; kx < 3; ++kx) {
            int ww = wq + kx - 1; if (ww < 0 || ww > 63) continue;
            acc += xz[(size_t)(b * 4096 + hh * 64 + ww) * 256 + d] * w[d * 9 + ky * 3 + kx];
        }
    }
    float v = f_silu(acc);
    size_t row = (size_t)b * 4096;
    xcpm[(row + p) * 128 + d] = v;
    Ahw [(row + p) * 128 + d] = (half_t)v;
    Awh [(row + (size_t)(wq * 64 + h)) * 128 + d] = (half_t)v;
}

// =============== selective scan: 16 blocks (b,k) x 512 threads (d x 4 state groups) ======
__global__ void k_scan(const float* __restrict__ G,    // [4][16384][48]
                       const float* __restrict__ xcpm, // [16384][128]
                       const float* __restrict__ dtw,  // [4][128][4]
                       const float* __restrict__ dtb,  // [4][128]
                       const float* __restrict__ Alog, // [512][16]
                       float* __restrict__ ys)         // [16][4096][128]
{
    int tid = threadIdx.x;          // 0..511
    int g = tid & 3, d = tid >> 2;  // 4 states per lane
    int bk = blockIdx.x; int b = bk >> 2, k = bk & 3;
    bool rev = (k & 2) != 0, wh = (k & 1) != 0;
    float A[4], hst[4];
    #pragma unroll
    for (int j = 0; j < 4; ++j) {
        A[j] = -__expf(Alog[(size_t)(k * 128 + d) * 16 + g * 4 + j]);
        hst[j] = 0.f;
    }
    float w0 = dtw[(k * 128 + d) * 4 + 0], w1 = dtw[(k * 128 + d) * 4 + 1];
    float w2 = dtw[(k * 128 + d) * 4 + 2], w3 = dtw[(k * 128 + d) * 4 + 3];
    float db = dtb[k * 128 + d];
    const float* Gk = G + ((size_t)k * 16384 + (size_t)b * 4096) * 48;
    const float* xb = xcpm + (size_t)b * 4096 * 128;
    float* yb = ys + (size_t)bk * 4096 * 128;
    for (int s = 0; s < 4096; ++s) {
        int p = rev ? 4095 - s : s;
        const float* gr = Gk + (size_t)p * 48;      // wave-uniform broadcast loads
        int phys = wh ? ((p & 63) * 64 + (p >> 6)) : p;
        float u = xb[(size_t)phys * 128 + d];
        float delta = f_softplus(fmaf(gr[0], w0, fmaf(gr[1], w1,
                                 fmaf(gr[2], w2, fmaf(gr[3], w3, db)))));
        float du = delta * u;
        float y = 0.f;
        #pragma unroll
        for (int j = 0; j < 4; ++j) {
            hst[j] = fmaf(__expf(delta * A[j]), hst[j], du * gr[4 + g * 4 + j]);
            y = fmaf(hst[j], gr[20 + g * 4 + j], y);
        }
        y += __shfl_xor(y, 1, 32);
        y += __shfl_xor(y, 2, 32);
        if (g == 0) yb[(size_t)s * 128 + d] = y;
    }
}

// combine 4 directions + D*u, out_norm LN, * silu(z) -> yact f16 (block per pixel)
__global__ void k_combine(const float* __restrict__ ys, const float* __restrict__ xcpm,
                          const float* __restrict__ Ds, const float* __restrict__ xz,
                          const float* __restrict__ g, const float* __restrict__ bb,
                          half_t* __restrict__ yact)
{
    __shared__ float red[128];
    int m = blockIdx.x; int d = threadIdx.x;
    int b = m >> 12; int p = m & 4095; int h = p >> 6, wq = p & 63;
    int pwh = wq * 64 + h;
    float y = ys[((size_t)(b * 4 + 0) * 4096 + p)          * 128 + d]
            + ys[((size_t)(b * 4 + 1) * 4096 + pwh)        * 128 + d]
            + ys[((size_t)(b * 4 + 2) * 4096 + (4095 - p)) * 128 + d]
            + ys[((size_t)(b * 4 + 3) * 4096 + (4095 - pwh)) * 128 + d];
    float u = xcpm[(size_t)m * 128 + d];
    y += (Ds[d] + Ds[128 + d] + Ds[256 + d] + Ds[384 + d]) * u;
    red[d] = y; __syncthreads();
    for (int o = 64; o > 0; o >>= 1) { if (d < o) red[d] += red[d + o]; __syncthreads(); }
    float mu = red[0] * (1.f / 128.f); __syncthreads();
    float dv = y - mu;
    red[d] = dv * dv; __syncthreads();
    for (int o = 64; o > 0; o >>= 1) { if (d < o) red[d] += red[d + o]; __syncthreads(); }
    float rs = rsqrtf(red[0] * (1.f / 128.f) + 1e-5f);
    float yn = dv * rs * g[d] + bb[d];
    float z = xz[(size_t)m * 256 + 128 + d];
    yact[(size_t)m * 128 + d] = (half_t)(yn * f_silu(z));
}

// x1 branch -> xcat channels [0,64) (+ inp)
__global__ void k_x1_final(const float* __restrict__ x1, const float* __restrict__ yproj,
                           const float* __restrict__ inp, float* __restrict__ xcat)
{
    int t = blockIdx.x * blockDim.x + threadIdx.x;   // (b*64+c)*4096 + p
    if (t >= 4 * 64 * 4096) return;
    int p = t & 4095; int c = (t >> 12) & 63; int b = t >> 18;
    size_t m = (size_t)b * 4096 + p;
    float v = x1[m * 64 + c] + yproj[m * 64 + c];
    size_t o = (size_t)(b * 128 + c) * 4096 + p;
    xcat[o] = v + inp[o];
}

// =============== FFT branch (ortho, 64-point DFTs via twiddle table) ===============
__global__ void k_twid(float* __restrict__ tw)
{
    int j = threadIdx.x;
    if (j < 64) {
        float th = 6.283185307179586f * (float)j / 64.f;
        tw[2 * j] = cosf(th); tw[2 * j + 1] = sinf(th);
    }
}

__global__ void k_rfft_w(const float* __restrict__ x2p, const float* __restrict__ tw,
                         float* __restrict__ Xr, float* __restrict__ Xi)
{
    int t = blockIdx.x * blockDim.x + threadIdx.x;   // ((b*64+c)*64+h)*33 + wf
    if (t >= 4 * 64 * 64 * 33) return;
    int wf = t % 33; int rest = t / 33;
    const float* row = x2p + (size_t)rest * 64;
    float ar = 0.f, ai = 0.f;
    for (int w = 0; w < 64; ++w) {
        int idx = (w * wf) & 63;
        float v = row[w];
        ar += v * tw[2 * idx]; ai -= v * tw[2 * idx + 1];
    }
    Xr[t] = ar * 0.125f; Xi[t] = ai * 0.125f;
}

__global__ void k_fft_h(const float* __restrict__ Xr, const float* __restrict__ Xi,
                        const float* __restrict__ tw,
                        float* __restrict__ Yr, float* __restrict__ Yi)
{
    int t = blockIdx.x * blockDim.x + threadIdx.x;   // bc*2112 + hf*33 + wf
    if (t >= 4 * 64 * 64 * 33) return;
    int wf = t % 33; int hf = (t / 33) & 63; int bc = t / 2112;
    size_t base = (size_t)bc * 2112 + wf;
    float ar = 0.f, ai = 0.f;
    for (int hh = 0; hh < 64; ++hh) {
        int idx = (hh * hf) & 63;
        float xr = Xr[base + hh * 33], xi = Xi[base + hh * 33];
        float cc = tw[2 * idx], ss = tw[2 * idx + 1];
        ar += xr * cc + xi * ss;     // * e^{-i th}
        ai += xi * cc - xr * ss;
    }
    Yr[t] = ar * 0.125f; Yi[t] = ai * 0.125f;
}

__global__ void k_pack_fft(const float* __restrict__ Xr, const float* __restrict__ Xi,
                           half_t* __restrict__ Afft)
{
    int t = blockIdx.x * blockDim.x + threadIdx.x;   // m*128 + c ; m=(b,hf,wf)
    if (t >= 8448 * 128) return;
    int c = t & 127; int m = t >> 7;
    int b = m / 2112; int q = m % 2112;
    float v = (c < 64) ? Xr[(size_t)(b * 64 + c) * 2112 + q]
                       : Xi[(size_t)(b * 64 + (c - 64)) * 2112 + q];
    Afft[t] = (half_t)v;
}

// BN + exact GELU + complex multiply with bilinear-upsampled weight
__global__ void k_fft_post(const float* __restrict__ Hc,
                           const float* __restrict__ bg, const float* __restrict__ bb,
                           const float* __restrict__ bm, const float* __restrict__ bv,
                           const float* __restrict__ cw,
                           float* __restrict__ Yr, float* __restrict__ Yi)
{
    int t = blockIdx.x * blockDim.x + threadIdx.x;   // (b*2112+q)*64 + c
    if (t >= 4 * 2112 * 64) return;
    int c = t & 63; int m = t >> 6; int q = m % 2112; int b = m / 2112;
    size_t row = (size_t)(b * 2112 + q) * 128;
    float hr = Hc[row + c], hi = Hc[row + 64 + c];
    hr = (hr - bm[c])      * rsqrtf(bv[c]      + 1e-5f) * bg[c]      + bb[c];
    hi = (hi - bm[64 + c]) * rsqrtf(bv[64 + c] + 1e-5f) * bg[64 + c] + bb[64 + c];
    hr = f_gelu(hr); hi = f_gelu(hi);
    int hf = q / 33, wf = q % 33;
    float yv = hf * (15.f / 63.f), xv = wf * (15.f / 32.f);
    int y0 = (int)floorf(yv); int y1 = y0 + 1; if (y1 > 15) y1 = 15;
    int x0 = (int)floorf(xv); int x1 = x0 + 1; if (x1 > 15) x1 = 15;
    float wy = yv - (float)y0, wx = xv - (float)x0;
    const float* p0 = cw + (size_t)c * 256;          // plane 0
    const float* p1 = cw + 16384 + (size_t)c * 256;  // plane 1
    float wr = p0[y0*16+x0]*(1-wy)*(1-wx) + p0[y0*16+x1]*(1-wy)*wx
             + p0[y1*16+x0]*wy*(1-wx)     + p0[y1*16+x1]*wy*wx;
    float wi = p1[y0*16+x0]*(1-wy)*(1-wx) + p1[y0*16+x1]*(1-wy)*wx
             + p1[y1*16+x0]*wy*(1-wx)     + p1[y1*16+x1]*wy*wx;
    size_t o = (size_t)(b * 64 + c) * 2112 + q;
    Yr[o] = hr * wr - hi * wi;
    Yi[o] = hr * wi + hi * wr;
}

__global__ void k_ifft_h(const float* __restrict__ Yr, const float* __restrict__ Yi,
                         const float* __restrict__ tw,
                         float* __restrict__ Zr, float* __restrict__ Zi)
{
    int t = blockIdx.x * blockDim.x + threadIdx.x;   // bc*2112 + h*33 + wf
    if (t >= 4 * 64 * 64 * 33) return;
    int wf = t % 33; int h = (t / 33) & 63; int bc = t / 2112;
    size_t base = (size_t)bc * 2112 + wf;
    float ar = 0.f, ai = 0.f;
    for (int hf = 0; hf < 64; ++hf) {
        int idx = (h * hf) & 63;
        float xr = Yr[base + hf * 33], xi = Yi[base + hf * 33];
        float cc = tw[2 * idx], ss = tw[2 * idx + 1];
        ar += xr * cc - xi * ss;     // * e^{+i th}
        ai += xr * ss + xi * cc;
    }
    Zr[t] = ar * 0.125f; Zi[t] = ai * 0.125f;
}

__global__ void k_irfft_w(const float* __restrict__ Zr, const float* __restrict__ Zi,
                          const float* __restrict__ tw, const float* __restrict__ inp,
                          float* __restrict__ xcat)
{
    int t = blockIdx.x * blockDim.x + threadIdx.x;   // bc*4096 + h*64 + w
    if (t >= 4 * 64 * 4096) return;
    int w = t & 63; int h = (t >> 6) & 63; int bc = t >> 12;
    size_t base = (size_t)bc * 2112 + (size_t)h * 33;
    float acc = Zr[base] + ((w & 1) ? -Zr[base + 32] : Zr[base + 32]);
    for (int kk = 1; kk < 32; ++kk) {
        int idx = (kk * w) & 63;
        acc += 2.f * (Zr[base + kk] * tw[2 * idx] - Zi[base + kk] * tw[2 * idx + 1]);
    }
    acc *= 0.125f;
    int b = bc >> 6, c = bc & 63;
    size_t o = (size_t)(b * 128 + 64 + c) * 4096 + (size_t)(h * 64 + w);
    xcat[o] = acc + inp[o];
}

// =============== tail: norm2 LN, dw 3x3, pw, residual, prj ===============
__global__ void k_ln2(const float* __restrict__ xcat, const float* __restrict__ g,
                      const float* __restrict__ bb, float* __restrict__ xm)
{
    __shared__ float red[128];
    int m = blockIdx.x; int d = threadIdx.x;
    int b = m >> 12; int p = m & 4095;
    size_t idx = (size_t)(b * 128 + d) * 4096 + p;
    float v = xcat[idx];
    red[d] = v; __syncthreads();
    for (int o = 64; o > 0; o >>= 1) { if (d < o) red[d] += red[d + o]; __syncthreads(); }
    float mu = red[0] * (1.f / 128.f); __syncthreads();
    float dv = v - mu;
    red[d] = dv * dv; __syncthreads();
    for (int o = 64; o > 0; o >>= 1) { if (d < o) red[d] += red[d + o]; __syncthreads(); }
    float rs = rsqrtf(red[0] * (1.f / 128.f) + 1e-5f);
    xm[idx] = dv * rs * g[d] + bb[d];
}

__global__ void k_dw2(const float* __restrict__ xm, const float* __restrict__ w,
                      const float* __restrict__ bias, half_t* __restrict__ Amlp)
{
    int t = blockIdx.x * blockDim.x + threadIdx.x;   // (b*128+c)*4096 + p
    if (t >= 4 * 128 * 4096) return;
    int p = t & 4095; int c = (t >> 12) & 127; int b = t >> 19;
    int h = p >> 6, wq = p & 63;
    const float* plane = xm + (size_t)(b * 128 + c) * 4096;
    float acc = bias[c];
    #pragma unroll
    for (int ky = 0; ky < 3; ++ky) {
        int hh = h + ky - 1; if (hh < 0 || hh > 63) continue;
        #pragma unroll
        for (int kx = 0; kx < 3; ++kx) {
            int ww = wq + kx - 1; if (ww < 0 || ww > 63) continue;
            acc += plane[hh * 64 + ww] * w[c * 9 + ky * 3 + kx];
        }
    }
    Amlp[((size_t)b * 4096 + p) * 128 + c] = (half_t)acc;
}

__global__ void k_add_pack(const float* __restrict__ P, const float* __restrict__ xcat,
                           half_t* __restrict__ Aprj)
{
    int t = blockIdx.x * blockDim.x + threadIdx.x;   // m*128 + c
    if (t >= 16384 * 128) return;
    int c = t & 127; int m = t >> 7; int b = m >> 12; int p = m & 4095;
    Aprj[t] = (half_t)(P[t] + xcat[(size_t)(b * 128 + c) * 4096 + p]);
}

__global__ void k_out_tr(const float* __restrict__ O, float* __restrict__ out)
{
    int t = blockIdx.x * blockDim.x + threadIdx.x;   // m*128 + c -> NCHW
    if (t >= 16384 * 128) return;
    int c = t & 127; int m = t >> 7; int b = m >> 12; int p = m & 4095;
    out[(size_t)(b * 128 + c) * 4096 + p] = O[t];
}

// ====================================================================================
static inline int cdiv(int a, int b) { return (a + b - 1) / b; }

static void gemm(hipStream_t s, const half_t* A, const half_t* Wt, const float* bias,
                 float* C, int M, int N, int K)
{
    int ntiles = (M >> 4) * (N >> 4);
    k_gemm<<<cdiv(ntiles, 8), 256, 0, s>>>(A, Wt, bias, C, N, K, ntiles);
}

extern "C" void kernel_launch(void* const* d_in, const int* in_sizes, int n_in,
                              void* d_out, int out_size, void* d_ws, size_t ws_size,
                              hipStream_t stream)
{
    (void)in_sizes; (void)n_in; (void)out_size; (void)ws_size;
    const float* x        = (const float*)d_in[0];
    const float* norm1_g  = (const float*)d_in[1];
    const float* norm1_b  = (const float*)d_in[2];
    const float* pe_w     = (const float*)d_in[3];
    const float* pe_b     = (const float*)d_in[4];
    const float* pe_ln_g  = (const float*)d_in[5];
    const float* pe_ln_b  = (const float*)d_in[6];
    const float* vss_ln_g = (const float*)d_in[7];
    const float* vss_ln_b = (const float*)d_in[8];
    const float* in_proj_w= (const float*)d_in[9];
    const float* cdw_w    = (const float*)d_in[10];
    const float* cdw_b    = (const float*)d_in[11];
    const float* x_proj_w = (const float*)d_in[12];
    const float* dt_w     = (const float*)d_in[13];
    const float* dt_b     = (const float*)d_in[14];
    const float* A_logs   = (const float*)d_in[15];
    const float* Ds       = (const float*)d_in[16];
    const float* onorm_g  = (const float*)d_in[17];
    const float* onorm_b  = (const float*)d_in[18];
    const float* oproj_w  = (const float*)d_in[19];
    const float* fconv_w  = (const float*)d_in[20];
    const float* fconv_b  = (const float*)d_in[21];
    const float* fbn_g    = (const float*)d_in[22];
    const float* fbn_b    = (const float*)d_in[23];
    const float* fbn_m    = (const float*)d_in[24];
    const float* fbn_v    = (const float*)d_in[25];
    const float* cweight  = (const float*)d_in[26];
    const float* norm2_g  = (const float*)d_in[27];
    const float* norm2_b  = (const float*)d_in[28];
    const float* mdw_w    = (const float*)d_in[29];
    const float* mdw_b    = (const float*)d_in[30];
    const float* mpw_w    = (const float*)d_in[31];
    const float* mpw_b    = (const float*)d_in[32];
    const float* prj_w    = (const float*)d_in[33];
    const float* prj_b    = (const float*)d_in[34];
    float* out = (float*)d_out;

    char* ws = (char*)d_ws;
    size_t off = 0;
    auto alloc = [&](size_t bytes) -> char* {
        char* p = ws + off; off = (off + bytes + 255) & ~(size_t)255; return p;
    };
    float*  mean  = (float*) alloc(65536u * 4);
    float*  rstd  = (float*) alloc(65536u * 4);
    float*  inp   = (float*) alloc(2097152u * 4);
    float*  xn1   = (float*) alloc(4194304u * 4);
    float*  x2p   = (float*) alloc(1048576u * 4);
    half_t* Ape   = (half_t*)alloc(16384u * 256 * 2);
    half_t* Wpe   = (half_t*)alloc(64u * 256 * 2);
    float*  peout = (float*) alloc(16384u * 64 * 4);
    float*  x1    = (float*) alloc(16384u * 64 * 4);
    half_t* x1n   = (half_t*)alloc(16384u * 64 * 2);
    half_t* Wip   = (half_t*)alloc(256u * 64 * 2);
    float*  xz    = (float*) alloc(16384u * 256 * 4);
    float*  xcpm  = (float*) alloc(16384u * 128 * 4);
    half_t* Ahw   = (half_t*)alloc(16384u * 128 * 2);
    half_t* Awh   = (half_t*)alloc(16384u * 128 * 2);
    half_t* Wxp   = (half_t*)alloc(4u * 48 * 128 * 2);
    float*  G     = (float*) alloc(4u * 16384 * 48 * 4);
    float*  ys    = (float*) alloc(16u * 4096 * 128 * 4);
    half_t* yact  = (half_t*)alloc(16384u * 128 * 2);
    half_t* Wop   = (half_t*)alloc(64u * 128 * 2);
    float*  yproj = (float*) alloc(16384u * 64 * 4);
    float*  xcat  = (float*) alloc(2097152u * 4);
    float*  twid  = (float*) alloc(128u * 4);
    float*  xfwr  = (float*) alloc(540672u * 4);
    float*  xfwi  = (float*) alloc(540672u * 4);
    float*  xfr   = (float*) alloc(540672u * 4);
    float*  xfi   = (float*) alloc(540672u * 4);
    half_t* Afft  = (half_t*)alloc(8448u * 128 * 2);
    half_t* Wfft  = (half_t*)alloc(128u * 128 * 2);
    float*  Hc    = (float*) alloc(8448u * 128 * 4);
    float*  Cr    = (float*) alloc(540672u * 4);
    float*  Ci_   = (float*) alloc(540672u * 4);
    float*  Zr    = (float*) alloc(540672u * 4);
    float*  Zi    = (float*) alloc(540672u * 4);
    float*  xm    = (float*) alloc(2097152u * 4);
    half_t* Amlp  = (half_t*)alloc(16384u * 128 * 2);
    half_t* Wpw   = (half_t*)alloc(128u * 128 * 2);
    float*  Pbuf  = (float*) alloc(16384u * 128 * 4);
    half_t* Aprj  = (half_t*)alloc(16384u * 128 * 2);
    half_t* Wprj  = (half_t*)alloc(128u * 128 * 2);
    float*  Obuf  = (float*) alloc(16384u * 128 * 4);

    const int TB = 256;
    // weight conversions + twiddle table
    k_cvt<<<cdiv(16384, TB), TB, 0, stream>>>(pe_w, Wpe, 16384);
    k_cvt<<<cdiv(16384, TB), TB, 0, stream>>>(in_proj_w, Wip, 16384);
    k_cvt<<<cdiv(8192,  TB), TB, 0, stream>>>(oproj_w, Wop, 8192);
    k_cvt<<<cdiv(16384, TB), TB, 0, stream>>>(fconv_w, Wfft, 16384);
    k_cvt<<<cdiv(16384, TB), TB, 0, stream>>>(mpw_w, Wpw, 16384);
    k_cvt<<<cdiv(16384, TB), TB, 0, stream>>>(prj_w, Wprj, 16384);
    k_pad_xpw<<<cdiv(24576, TB), TB, 0, stream>>>(x_proj_w, Wxp);
    k_twid<<<1, 64, 0, stream>>>(twid);

    // stage 1: LN + pools
    k_ln_stats<<<cdiv(65536, TB), TB, 0, stream>>>(x, mean, rstd);
    k_prep1<<<cdiv(2097152, TB), TB, 0, stream>>>(x, mean, rstd, norm1_g, norm1_b,
                                                  inp, xn1, x2p);
    // patch embed (im2col GEMM) + double LN
    k_pack_peA<<<cdiv(1048576, TB), TB, 0, stream>>>(xn1, Ape);
    gemm(stream, Ape, Wpe, pe_b, peout, 16384, 64, 256);
    k_pe_ln<<<cdiv(16384, TB), TB, 0, stream>>>(peout, pe_ln_g, pe_ln_b,
                                                vss_ln_g, vss_ln_b, x1, x1n);
    // SS2D
    gemm(stream, x1n, Wip, nullptr, xz, 16384, 256, 64);
    k_dw_silu<<<cdiv(2097152, TB), TB, 0, stream>>>(xz, cdw_w, cdw_b, xcpm, Ahw, Awh);
    gemm(stream, Ahw, Wxp + 0 * 48 * 128, nullptr, G + (size_t)0 * 16384 * 48, 16384, 48, 128);
    gemm(stream, Awh, Wxp + 1 * 48 * 128, nullptr, G + (size_t)1 * 16384 * 48, 16384, 48, 128);
    gemm(stream, Ahw, Wxp + 2 * 48 * 128, nullptr, G + (size_t)2 * 16384 * 48, 16384, 48, 128);
    gemm(stream, Awh, Wxp + 3 * 48 * 128, nullptr, G + (size_t)3 * 16384 * 48, 16384, 48, 128);
    k_scan<<<16, 512, 0, stream>>>(G, xcpm, dt_w, dt_b, A_logs, ys);
    k_combine<<<16384, 128, 0, stream>>>(ys, xcpm, Ds, xz, onorm_g, onorm_b, yact);
    gemm(stream, yact, Wop, nullptr, yproj, 16384, 64, 128);
    k_x1_final<<<cdiv(1048576, TB), TB, 0, stream>>>(x1, yproj, inp, xcat);
    // FFT branch
    k_rfft_w<<<cdiv(540672, TB), TB, 0, stream>>>(x2p, twid, xfwr, xfwi);
    k_fft_h<<<cdiv(540672, TB), TB, 0, stream>>>(xfwr, xfwi, twid, xfr, xfi);
    k_pack_fft<<<cdiv(1081344, TB), TB, 0, stream>>>(xfr, xfi, Afft);
    gemm(stream, Afft, Wfft, fconv_b, Hc, 8448, 128, 128);
    k_fft_post<<<cdiv(540672, TB), TB, 0, stream>>>(Hc, fbn_g, fbn_b, fbn_m, fbn_v,
                                                    cweight, Cr, Ci_);
    k_ifft_h<<<cdiv(540672, TB), TB, 0, stream>>>(Cr, Ci_, twid, Zr, Zi);
    k_irfft_w<<<cdiv(1048576, TB), TB, 0, stream>>>(Zr, Zi, twid, inp, xcat);
    // tail
    k_ln2<<<16384, 128, 0, stream>>>(xcat, norm2_g, norm2_b, xm);
    k_dw2<<<cdiv(2097152, TB), TB, 0, stream>>>(xm, mdw_w, mdw_b, Amlp);
    gemm(stream, Amlp, Wpw, mpw_b, Pbuf, 16384, 128, 128);
    k_add_pack<<<cdiv(2097152, TB), TB, 0, stream>>>(Pbuf, xcat, Aprj);
    gemm(stream, Aprj, Wprj, prj_b, Obuf, 16384, 128, 128);
    k_out_tr<<<cdiv(2097152, TB), TB, 0, stream>>>(Obuf, out);
}
